// DecoderModel_4191888081009
// MI455X (gfx1250) — matrix operand order repaired and checked by reading, not compiled, not run on hardware
//
#include <hip/hip_runtime.h>
#include <hip/hip_bf16.h>
#include <stdint.h>

typedef __attribute__((ext_vector_type(16))) __bf16 v16bf;
typedef __attribute__((ext_vector_type(8)))  float  v8f;
typedef __attribute__((ext_vector_type(4)))  int    v4i;

#define DEVFN __device__ __forceinline__

#if defined(__HIP_DEVICE_COMPILE__) && __has_builtin(__builtin_amdgcn_global_load_async_to_lds_b128)
#define HAS_ASYNC_LDS 1
#else
#define HAS_ASYNC_LDS 0
#endif

namespace {

constexpr int V_   = 32000;
constexpr int D_   = 512;
constexpr int H_   = 8;
constexpr int L_   = 6;
constexpr int FF_  = 2048;
constexpr int B_   = 2;
constexpr int S_   = 1024;
constexpr int DH_  = D_ / H_;   // 64
constexpr int WIN_ = 64;
constexpr int M_   = B_ * S_;   // 2048 token rows
constexpr float LN_EPS = 1e-5f;

union FragB16 { v16bf v; uint4 q[2]; };

DEVFN v8f wmma_bf16(v16bf a, v16bf b, v8f c) {
  // v_wmma_f32_16x16x32_bf16  D = A(16x32) * B(32x16) + C
  return __builtin_amdgcn_wmma_f32_16x16x32_bf16(
      /*neg_a=*/false, a, /*neg_b=*/false, b,
      /*c_mod=*/(short)0, c, /*reuse_a=*/false, /*reuse_b=*/false);
}

#if HAS_ASYNC_LDS
typedef __attribute__((address_space(1))) v4i* as1_v4i_p;
typedef __attribute__((address_space(3))) v4i* as3_v4i_p;

DEVFN void async_b128(const __bf16* g, __bf16* l) {
  // GLOBAL_LOAD_ASYNC_TO_LDS_B128: per-lane 16B global->LDS, tracked by ASYNCcnt.
  // Generic->as(1)/as(3) via integer casts (low 32 bits of a generic LDS address
  // are the as(3) offset per the flat-aperture rules).
  __builtin_amdgcn_global_load_async_to_lds_b128(
      (as1_v4i_p)(uintptr_t)g,
      (as3_v4i_p)(unsigned int)(uintptr_t)l,
      0, 0);
}
DEVFN void wait_async0() {
#if __has_builtin(__builtin_amdgcn_s_wait_asynccnt)
  __builtin_amdgcn_s_wait_asynccnt(0);
#else
  asm volatile("s_wait_asynccnt 0" ::: "memory");
#endif
}
#endif

// A fragment (16x32 bf16, MxK), lane holds row (lane&15):
//   lanes 0-15: K = 0..15;  lanes 16-31: K = 16..31   (contiguous per lane)
DEVFN v16bf load_frag_a_rowptr(const __bf16* rowp, int lane) {
  int kh = (lane >> 4) * 16;
  FragB16 f;
  f.q[0] = *(const uint4*)(rowp + kh);
  f.q[1] = *(const uint4*)(rowp + kh + 8);
  return f.v;
}
// B fragment (32x16 bf16, KxN), lane holds column (lane&15):
//   lanes 0-15: K = 0..15;  lanes 16-31: K = 16..31   (contiguous per lane)
DEVFN v16bf load_frag_b_rowptr(const __bf16* rowp, int lane) {
  int koff = (lane >> 4) * 16;
  FragB16 f;
  f.q[0] = *(const uint4*)(rowp + koff);
  f.q[1] = *(const uint4*)(rowp + koff + 8);
  return f.v;
}
DEVFN v16bf load_frag_a(const __bf16* base, int lane, int ld) {
  return load_frag_a_rowptr(base + (lane & 15) * ld, lane);
}
DEVFN v16bf load_frag_b(const __bf16* base, int lane, int ld) {
  return load_frag_b_rowptr(base + (lane & 15) * ld, lane);
}

DEVFN float halfwave_max(float v) {
  #pragma unroll
  for (int off = 8; off >= 1; off >>= 1) v = fmaxf(v, __shfl_xor(v, off, 32));
  return v;
}
DEVFN float halfwave_sum(float v) {
  #pragma unroll
  for (int off = 8; off >= 1; off >>= 1) v += __shfl_xor(v, off, 32);
  return v;
}

// ---------------------------------------------------------------- cvt f32->bf16
__global__ __launch_bounds__(256)
void cvt_kernel(const float* __restrict__ in, __bf16* __restrict__ out, long long n) {
  long long i = (long long)blockIdx.x * 256 + threadIdx.x;
  long long stride = (long long)gridDim.x * 256;
  for (; i < n; i += stride) out[i] = (__bf16)in[i];
}

// ---------------------------------------------------------------- embedding
__global__ __launch_bounds__(256)
void embed_kernel(const int* __restrict__ ids, const float* __restrict__ emb,
                  const float* __restrict__ pos, float* __restrict__ x,
                  __bf16* __restrict__ xb) {
  int t = blockIdx.x;
  int s = t % S_;
  int id = ids[t];
  size_t eb = (size_t)id * D_, pb = (size_t)s * D_, ob = (size_t)t * D_;
  for (int d = threadIdx.x; d < D_; d += 256) {
    float v = emb[eb + d] + pos[pb + d];
    x[ob + d] = v;
    xb[ob + d] = (__bf16)v;
  }
}

// ---------------------------------------------------------------- GEMM (bf16 WMMA)
// C[M,N] = A[M,K] @ W[N,K]^T + bias ; optional relu ; C (f32) and/or Cb (bf16).
// Double-buffered LDS staging via GLOBAL_LOAD_ASYNC_TO_LDS_B128 when available.
template <bool HASC, bool HASCB>
__global__ __launch_bounds__(256, 2)
void gemm_bf16_kernel(const __bf16* __restrict__ A, const __bf16* __restrict__ W,
                      const float* __restrict__ bias, float* __restrict__ C,
                      __bf16* __restrict__ Cb, int M, int N, int K, int relu) {
  constexpr int LDT = 40;  // padded K-stride (elems) in LDS to stagger banks
#if HAS_ASYNC_LDS
  constexpr int NBUF = 2;
#else
  constexpr int NBUF = 1;
#endif
  __shared__ __align__(16) __bf16 As[NBUF][128 * LDT];
  __shared__ __align__(16) __bf16 Bs[NBUF][128 * LDT];

  const int n0 = blockIdx.x * 128;
  const int m0 = blockIdx.y * 128;
  const int tid  = threadIdx.x;
  const int lane = tid & 31;
  const int wave = tid >> 5;
  const int wm = (wave & 3) * 32;   // 4 wave-rows  x 32 M
  const int wn = (wave >> 2) * 64;  // 2 wave-cols  x 64 N

  const int lrow  = tid >> 1;        // 0..127 tile row for staging
  const int lhalf = (tid & 1) * 16;  // first/second 16 elems of the 32-K slab

  v8f zero = {};
  v8f acc[2][4];
  #pragma unroll
  for (int i = 0; i < 2; ++i)
    #pragma unroll
    for (int j = 0; j < 4; ++j) acc[i][j] = zero;

#if HAS_ASYNC_LDS
  auto issue = [&](int buf, int k0) {
    const __bf16* ga = A + (size_t)(m0 + lrow) * K + k0 + lhalf;
    const __bf16* gb = W + (size_t)(n0 + lrow) * K + k0 + lhalf;
    __bf16* la = &As[buf][lrow * LDT + lhalf];
    __bf16* lb = &Bs[buf][lrow * LDT + lhalf];
    async_b128(ga,     la);
    async_b128(ga + 8, la + 8);
    async_b128(gb,     lb);
    async_b128(gb + 8, lb + 8);
  };
  issue(0, 0);
#endif

  const int ksteps = K / 32;
  for (int ks = 0; ks < ksteps; ++ks) {
    const int k0 = ks * 32;
#if HAS_ASYNC_LDS
    const int buf = ks & 1;
    wait_async0();       // our slab's async writes to LDS have landed
    __syncthreads();     // everyone's have; previous slab fully consumed
    if (ks + 1 < ksteps) issue(buf ^ 1, k0 + 32);  // overlap next slab with math
#else
    const int buf = 0;
    const uint4* ga = (const uint4*)(A + (size_t)(m0 + lrow) * K + k0 + lhalf);
    const uint4* gb = (const uint4*)(W + (size_t)(n0 + lrow) * K + k0 + lhalf);
    uint4 a0 = ga[0], a1 = ga[1];
    uint4 b0 = gb[0], b1 = gb[1];
    if (k0 + 32 < K) {  // global_prefetch_b8 of next K slab
      __builtin_prefetch(A + (size_t)(m0 + lrow) * K + k0 + 32 + lhalf, 0, 1);
      __builtin_prefetch(W + (size_t)(n0 + lrow) * K + k0 + 32 + lhalf, 0, 1);
    }
    __syncthreads();
    *(uint4*)(&As[0][lrow * LDT + lhalf])     = a0;
    *(uint4*)(&As[0][lrow * LDT + lhalf + 8]) = a1;
    *(uint4*)(&Bs[0][lrow * LDT + lhalf])     = b0;
    *(uint4*)(&Bs[0][lrow * LDT + lhalf + 8]) = b1;
    __syncthreads();
#endif

    v16bf af[2], bfr[4];
    #pragma unroll
    for (int i = 0; i < 2; ++i) af[i]  = load_frag_a(&As[buf][(wm + i * 16) * LDT], lane, LDT);
    #pragma unroll
    for (int j = 0; j < 4; ++j) bfr[j] = load_frag_b(&Bs[buf][(wn + j * 16) * LDT], lane, LDT);
    #pragma unroll
    for (int i = 0; i < 2; ++i)
      #pragma unroll
      for (int j = 0; j < 4; ++j)
        acc[i][j] = wmma_bf16(af[i], bfr[j], acc[i][j]);
  }

  // epilogue: C/D layout — VGPR r -> row (r | r+8 per half), lane&15 -> column
  #pragma unroll
  for (int j = 0; j < 4; ++j) {
    int n = n0 + wn + j * 16 + (lane & 15);
    float bn = bias ? bias[n] : 0.0f;
    #pragma unroll
    for (int i = 0; i < 2; ++i) {
      #pragma unroll
      for (int r = 0; r < 8; ++r) {
        int m = m0 + wm + i * 16 + r + ((lane >> 4) << 3);
        float v = acc[i][j][r] + bn;
        if (relu) v = fmaxf(v, 0.0f);
        size_t idx = (size_t)m * N + n;
        if (HASC)  C[idx]  = v;
        if (HASCB) Cb[idx] = (__bf16)v;
      }
    }
  }
}

// ---------------------------------------------------------------- attention
// One wave32 per (16-query tile, head, batch). Flash-style over 32-key blocks.
__global__ __launch_bounds__(32)
void attn_kernel(const __bf16* __restrict__ qkvb, const int* __restrict__ ids,
                 const int* __restrict__ pad_ptr, __bf16* __restrict__ outb) {
  __shared__ __align__(16) __bf16 pbuf[16 * 40];
  const int lane = threadIdx.x;
  const int q0 = blockIdx.x * 16;
  const int h  = blockIdx.y;
  const int b  = blockIdx.z;
  const int pad_id = *pad_ptr;
  const int win = (h + 1) * WIN_;
  const float scale = 0.125f;  // 1/sqrt(64)
  const size_t rs = 3 * D_;    // qkv row stride (elems)
  const int qhalf = (lane >> 4) << 3;

  // Q fragments (dh=64 -> two 32-wide K chunks)
  v16bf qf[2];
  {
    int q = q0 + (lane & 15);
    const __bf16* qrow = qkvb + (size_t)(b * S_ + q) * rs + h * DH_;
    #pragma unroll
    for (int c = 0; c < 2; ++c) qf[c] = load_frag_a_rowptr(qrow + c * 32, lane);
  }

  float m_i[8], l_i[8];
  #pragma unroll
  for (int r = 0; r < 8; ++r) { m_i[r] = -1e30f; l_i[r] = 0.0f; }
  v8f zero = {};
  v8f oacc[4];
  #pragma unroll
  for (int j = 0; j < 4; ++j) oacc[j] = zero;

  int lo = q0 - win; if (lo < 0) lo = 0; lo &= ~31;
  const int hi = q0 + 15;

  for (int kb = lo; kb <= hi; kb += 32) {
    // ---- scores: S[16q x 32k] as two 16x16 tiles
    v8f s0 = zero, s1 = zero;
    #pragma unroll
    for (int c = 0; c < 2; ++c) {
      int k0c = kb + (lane & 15);      int kc0 = k0c < S_ ? k0c : S_ - 1;
      int k1c = kb + 16 + (lane & 15); int kc1 = k1c < S_ ? k1c : S_ - 1;
      const __bf16* kp0 = qkvb + (size_t)(b * S_ + kc0) * rs + D_ + h * DH_ + c * 32;
      const __bf16* kp1 = qkvb + (size_t)(b * S_ + kc1) * rs + D_ + h * DH_ + c * 32;
      v16bf bk0 = load_frag_b_rowptr(kp0, lane);
      v16bf bk1 = load_frag_b_rowptr(kp1, lane);
      s0 = wmma_bf16(qf[c], bk0, s0);
      s1 = wmma_bf16(qf[c], bk1, s1);
    }

    // ---- mask + online softmax (rows live across 16-lane halves)
    int kcol0 = kb + (lane & 15), kcol1 = kcol0 + 16;
    bool padk0 = (kcol0 >= S_) || (ids[b * S_ + (kcol0 < S_ ? kcol0 : S_ - 1)] == pad_id);
    bool padk1 = (kcol1 >= S_) || (ids[b * S_ + (kcol1 < S_ ? kcol1 : S_ - 1)] == pad_id);
    float p0[8], p1[8];
    #pragma unroll
    for (int r = 0; r < 8; ++r) {
      int q = q0 + r + qhalf;
      float v0 = s0[r], v1 = s1[r];
      bool ok0 = (!padk0) && (kcol0 <= q) && (kcol0 >= q - win);
      bool ok1 = (!padk1) && (kcol1 <= q) && (kcol1 >= q - win);
      v0 = ok0 ? v0 * scale : -1e30f;
      v1 = ok1 ? v1 * scale : -1e30f;
      float bm = halfwave_max(fmaxf(v0, v1));
      float mn = fmaxf(m_i[r], bm);
      float alpha = __expf(m_i[r] - mn);                  // finite math, no NaN
      float e0 = (v0 < -1e29f) ? 0.0f : __expf(v0 - mn);
      float e1 = (v1 < -1e29f) ? 0.0f : __expf(v1 - mn);
      float rsum = halfwave_sum(e0 + e1);
      l_i[r] = l_i[r] * alpha + rsum;
      m_i[r] = mn;
      p0[r] = e0; p1[r] = e1;
      #pragma unroll
      for (int j = 0; j < 4; ++j) oacc[j][r] *= alpha;
    }

    // ---- stage P (C-layout) -> LDS -> A-fragment (same-wave LDS is in-order)
    asm volatile("" ::: "memory");
    #pragma unroll
    for (int r = 0; r < 8; ++r) {
      int qr = r + qhalf;
      pbuf[qr * 40 + (lane & 15)]      = (__bf16)p0[r];
      pbuf[qr * 40 + 16 + (lane & 15)] = (__bf16)p1[r];
    }
    asm volatile("" ::: "memory");
    v16bf pf = load_frag_a(pbuf, lane, 40);

    // ---- O += P @ V   (B fragment of V gathered: column = dh, K = key)
    #pragma unroll
    for (int j = 0; j < 4; ++j) {
      int d = j * 16 + (lane & 15);
      v16bf vv;
      #pragma unroll
      for (int i = 0; i < 16; ++i) {
        int key = kb + ((lane >> 4) << 4) + i;
        int kc = key < S_ ? key : S_ - 1;
        vv[i] = qkvb[(size_t)(b * S_ + kc) * rs + 2 * D_ + h * DH_ + d];
      }
      oacc[j] = wmma_bf16(pf, vv, oacc[j]);
    }
  }

  // ---- write O / l as bf16 into per-head columns of [M, D]
  #pragma unroll
  for (int j = 0; j < 4; ++j) {
    int d = h * DH_ + j * 16 + (lane & 15);
    #pragma unroll
    for (int r = 0; r < 8; ++r) {
      int q = q0 + r + qhalf;
      float v = oacc[j][r] / l_i[r];
      outb[(size_t)(b * S_ + q) * D_ + d] = (__bf16)v;
    }
  }
}

// ---------------------------------------------------------------- residual + LayerNorm
__global__ __launch_bounds__(256)
void addln_kernel(const float* __restrict__ x, const float* __restrict__ y,
                  const float* __restrict__ g, const float* __restrict__ be,
                  float* __restrict__ xout, __bf16* __restrict__ xbout) {
  __shared__ float ssum[256];
  __shared__ float ssq[256];
  const int t = blockIdx.x, tid = threadIdx.x;
  const size_t base = (size_t)t * D_;
  float a0 = x[base + tid]       + (y ? y[base + tid]       : 0.0f);
  float a1 = x[base + tid + 256] + (y ? y[base + tid + 256] : 0.0f);
  ssum[tid] = a0 + a1;
  ssq[tid]  = a0 * a0 + a1 * a1;
  __syncthreads();
  #pragma unroll
  for (int s = 128; s > 0; s >>= 1) {
    if (tid < s) { ssum[tid] += ssum[tid + s]; ssq[tid] += ssq[tid + s]; }
    __syncthreads();
  }
  float mu  = ssum[0] * (1.0f / D_);
  float var = ssq[0] * (1.0f / D_) - mu * mu;
  float inv = rsqrtf(var + LN_EPS);
  float o0 = (a0 - mu) * inv * g[tid]       + be[tid];
  float o1 = (a1 - mu) * inv * g[tid + 256] + be[tid + 256];
  if (xout)  { xout[base + tid] = o0;            xout[base + tid + 256] = o1; }
  if (xbout) { xbout[base + tid] = (__bf16)o0;   xbout[base + tid + 256] = (__bf16)o1; }
}

} // namespace

// ---------------------------------------------------------------- host driver
extern "C" void kernel_launch(void* const* d_in, const int* in_sizes, int n_in,
                              void* d_out, int out_size, void* d_ws, size_t ws_size,
                              hipStream_t stream) {
  const int*   ids  = (const int*)d_in[0];
  const int*   padp = (const int*)d_in[1];
  const float* emb  = (const float*)d_in[2];
  const float* pos  = (const float*)d_in[3];
  const float* Wqkv = (const float*)d_in[4];
  const float* bqkv = (const float*)d_in[5];
  const float* Wo   = (const float*)d_in[6];
  const float* bo   = (const float*)d_in[7];
  const float* W1   = (const float*)d_in[8];
  const float* b1   = (const float*)d_in[9];
  const float* W2   = (const float*)d_in[10];
  const float* b2   = (const float*)d_in[11];
  const float* g1   = (const float*)d_in[12];
  const float* be1  = (const float*)d_in[13];
  const float* g2   = (const float*)d_in[14];
  const float* be2  = (const float*)d_in[15];
  const float* gf   = (const float*)d_in[16];
  const float* bfv  = (const float*)d_in[17];
  (void)in_sizes; (void)n_in; (void)out_size; (void)ws_size;

  char* ws = (char*)d_ws;
  size_t off = 0;
  auto alloc = [&](size_t bytes) -> void* {
    off = (off + 255) & ~(size_t)255;
    void* p = ws + off;
    off += bytes;
    return p;
  };

  __bf16* embb  = (__bf16*)alloc((size_t)V_ * D_ * 2);
  __bf16* Wqkvb = (__bf16*)alloc((size_t)L_ * 3 * D_ * D_ * 2);
  __bf16* Wob   = (__bf16*)alloc((size_t)L_ * D_ * D_ * 2);
  __bf16* W1b   = (__bf16*)alloc((size_t)L_ * FF_ * D_ * 2);
  __bf16* W2b   = (__bf16*)alloc((size_t)L_ * D_ * FF_ * 2);
  float*  x     = (float*) alloc((size_t)M_ * D_ * 4);
  __bf16* xb    = (__bf16*)alloc((size_t)M_ * D_ * 2);
  __bf16* qkvb  = (__bf16*)alloc((size_t)M_ * 3 * D_ * 2);
  __bf16* attnb = (__bf16*)alloc((size_t)M_ * D_ * 2);
  float*  obuf  = (float*) alloc((size_t)M_ * D_ * 4);
  __bf16* ffhb  = (__bf16*)alloc((size_t)M_ * FF_ * 2);
  float*  ffo   = (float*) alloc((size_t)M_ * D_ * 4);
  __bf16* xfb   = (__bf16*)alloc((size_t)M_ * D_ * 2);

  auto cvt = [&](const float* in, __bf16* out, long long n) {
    long long blocks = (n + 255) / 256;
    if (blocks > 8192) blocks = 8192;
    cvt_kernel<<<(int)blocks, 256, 0, stream>>>(in, out, n);
  };
  // templated dispatch keeps the epilogue branch-free
  auto gemm = [&](const __bf16* A, const __bf16* Wm, const float* bias,
                  float* C, __bf16* Cb, int M, int N, int K, int relu) {
    dim3 grid(N / 128, M / 128);
    if (C && Cb)
      gemm_bf16_kernel<true, true><<<grid, 256, 0, stream>>>(A, Wm, bias, C, Cb, M, N, K, relu);
    else if (C)
      gemm_bf16_kernel<true, false><<<grid, 256, 0, stream>>>(A, Wm, bias, C, Cb, M, N, K, relu);
    else
      gemm_bf16_kernel<false, true><<<grid, 256, 0, stream>>>(A, Wm, bias, C, Cb, M, N, K, relu);
  };

  // one-time bf16 weight conversion (halves HBM traffic; L2-resident per layer)
  cvt(emb,  embb,  (long long)V_ * D_);
  cvt(Wqkv, Wqkvb, (long long)L_ * 3 * D_ * D_);
  cvt(Wo,   Wob,   (long long)L_ * D_ * D_);
  cvt(W1,   W1b,   (long long)L_ * FF_ * D_);
  cvt(W2,   W2b,   (long long)L_ * D_ * FF_);

  embed_kernel<<<M_, 256, 0, stream>>>(ids, emb, pos, x, xb);

  for (int l = 0; l < L_; ++l) {
    gemm(xb, Wqkvb + (size_t)l * 3 * D_ * D_, bqkv + (size_t)l * 3 * D_,
         nullptr, qkvb, M_, 3 * D_, D_, 0);
    attn_kernel<<<dim3(S_ / 16, H_, B_), 32, 0, stream>>>(qkvb, ids, padp, attnb);
    gemm(attnb, Wob + (size_t)l * D_ * D_, bo + (size_t)l * D_,
         obuf, nullptr, M_, D_, D_, 0);
    addln_kernel<<<M_, 256, 0, stream>>>(x, obuf, g1 + (size_t)l * D_,
                                         be1 + (size_t)l * D_, x, xb);
    gemm(xb, W1b + (size_t)l * FF_ * D_, b1 + (size_t)l * FF_,
         nullptr, ffhb, M_, FF_, D_, 1);
    gemm(ffhb, W2b + (size_t)l * D_ * FF_, b2 + (size_t)l * D_,
         ffo, nullptr, M_, D_, FF_, 0);
    addln_kernel<<<M_, 256, 0, stream>>>(x, ffo, g2 + (size_t)l * D_,
                                         be2 + (size_t)l * D_, x, xb);
  }

  addln_kernel<<<M_, 256, 0, stream>>>(x, nullptr, gf, bfv, nullptr, xfb);
  // tied LM head: [2048 x 512] @ [32000 x 512]^T  -> d_out (f32)
  gemm(xfb, embb, nullptr, (float*)d_out, nullptr, M_, V_, D_, 0);
}
